// Conv2_5D_Depth_63977832842009
// MI455X (gfx1250) — matrix operand order repaired and checked by reading, not compiled, not run on hardware
//
#include <hip/hip_runtime.h>
#include <hip/hip_bf16.h>

typedef __attribute__((ext_vector_type(16))) __bf16       v16bf;
typedef __attribute__((ext_vector_type(8)))  float        v8f;
typedef __attribute__((ext_vector_type(4)))  unsigned int v4u;
typedef __attribute__((ext_vector_type(4)))  unsigned int u32x4;
typedef __attribute__((ext_vector_type(8)))  int          i32x8;
typedef __attribute__((ext_vector_type(4)))  int          i32x4;

union BFrag { v16bf v; v4u q[2]; };

#define NIMG 8
#define CIN  128
#define OOUT 128
#define HH   64
#define WW   64
#define LTOT 4096                // HH*WW
#define KPP  9
#define NBR  3
#define NGRP (KPP * NBR)         // 27 (kp,branch) groups of 128 K each
#define KTOT (NGRP * CIN)        // 3456
#define H2   66                  // padded height
#define W2   66                  // padded width

#define TSTR 136                 // LDS row/col stride in elements (272 B, 16B-aligned,
                                 // 68-dword stride => conflict-light b128 access)
#define ABUF_BYTES (128 * TSTR * 2)   // 34816 per A buffer

// workspace layout (bytes)
#define WP_OFF 0
#define XP_OFF (OOUT * KTOT * 2)                        // 884736
#define MK_OFF (XP_OFF + NIMG * H2 * W2 * CIN * 2)      // 9805824
// total ws: MK_OFF + 3*8*9*4096 = 10690560 bytes (~10.2 MB)

#if __has_builtin(__builtin_amdgcn_tensor_load_to_lds) && \
    __has_builtin(__builtin_amdgcn_s_wait_tensorcnt)
#define USE_TDM 1
#else
#define USE_TDM 0
#endif

// ---------------------------------------------------------------------------
// 1) x (f32, NCHW) -> zero-padded channel-last bf16: xpad[n][y][x1][c]
//    (coalesced reads; scattered b16 writes -- one-time 25 MB transpose pass)
// ---------------------------------------------------------------------------
__global__ void pad_x_kernel(const float* __restrict__ x, __bf16* __restrict__ xpad) {
    int i = blockIdx.x * 256 + threadIdx.x;       // over (n, c, y, x1), x1 inner
    if (i >= NIMG * CIN * H2 * W2) return;
    int x1 = i % W2;
    int t  = i / W2;
    int y  = t % H2;
    int t2 = t / H2;
    int c  = t2 % CIN;
    int n  = t2 / CIN;
    float v = 0.0f;
    if (y >= 1 && y <= HH && x1 >= 1 && x1 <= WW)
        v = x[((n * CIN + c) * HH + (y - 1)) * WW + (x1 - 1)];
    xpad[(((size_t)n * H2 + y) * W2 + x1) * CIN + c] = (__bf16)v;
}

// ---------------------------------------------------------------------------
// 2) pack weights: Wp[o][(kp*3+b)*128 + c] = w_b[o][c][kp]  (bf16)
// ---------------------------------------------------------------------------
__global__ void pack_w_kernel(const float* __restrict__ w0, const float* __restrict__ w1,
                              const float* __restrict__ w2, __bf16* __restrict__ Wp) {
    int i = blockIdx.x * 256 + threadIdx.x;
    if (i >= OOUT * KTOT) return;
    int o  = i / KTOT;
    int kk = i - o * KTOT;
    int g  = kk >> 7;          // (kp*3 + b)
    int c  = kk & 127;
    int kp = g / 3;
    int b  = g - kp * 3;
    const float* w = (b == 0) ? w0 : ((b == 1) ? w1 : w2);
    Wp[i] = (__bf16)w[(o * CIN + c) * KPP + kp];
}

// ---------------------------------------------------------------------------
// 3) binary masks m[b][n][kp][l] as bytes (reference-faithful formulas)
// ---------------------------------------------------------------------------
__global__ void mask_kernel(const float* __restrict__ depth, const float* __restrict__ fx,
                            unsigned char* __restrict__ masks) {
    int i = blockIdx.x * 256 + threadIdx.x;   // n*LTOT + l, total 32768
    int n = i >> 12;
    int l = i & 4095;
    int oh = l >> 6, ow = l & 63;

    float dc[KPP];
#pragma unroll
    for (int kp = 0; kp < KPP; ++kp) {
        int ih = oh + kp / 3 - 1;
        int iw = ow + kp % 3 - 1;
        dc[kp] = ((unsigned)ih < (unsigned)HH && (unsigned)iw < (unsigned)WW)
                     ? depth[n * LTOT + ih * WW + iw] : 0.0f;
    }
    float cval   = (dc[4] != 0.0f) ? 1.0f : 0.0f;
    float center = dc[4] * cval;
    float grid   = center / fx[n];            // PIXEL_SIZE*DILATION == 1
    float halfg  = 0.5f * grid;

#pragma unroll
    for (int kp = 0; kp < KPP; ++kp) {
        float vd = (dc[kp] != 0.0f) ? cval : 0.0f;
        float dm = dc[kp] * vd;
        float m0 = (fabsf(dm - (center + grid)) <= halfg) ? 1.0f : 0.0f;
        float m1 = (fabsf(dm - center) <= halfg) ? 1.0f : 0.0f;
        m1 = fminf(m1 + 1.0f - vd, 1.0f);     // clip(m1 + 1 - valid, 0, 1)
        float m2 = (fabsf(dm - (center - grid)) <= halfg) ? 1.0f : 0.0f;
        masks[((0 * NIMG + n) * KPP + kp) * LTOT + l] = (unsigned char)(m0 != 0.0f);
        masks[((1 * NIMG + n) * KPP + kp) * LTOT + l] = (unsigned char)(m1 != 0.0f);
        masks[((2 * NIMG + n) * KPP + kp) * LTOT + l] = (unsigned char)(m2 != 0.0f);
    }
}

// ---------------------------------------------------------------------------
// TDM: DMA a 128(row) x 128(col) bf16 tile of Wp (row stride KTOT) into LDS,
// with LDS padding of 16 B per 256 B row (row stride 272 B == TSTR elements).
// D# per CDNA5 ISA 8.3/8.4: count=1, data_size=2B, pad_interval=64 dwords (5),
// pad_amount=4 dwords (3).
// ---------------------------------------------------------------------------
#if USE_TDM
__device__ __forceinline__ void tdm_load_a_tile(const __bf16* gsrc, unsigned lds_off) {
    unsigned long long ga = (unsigned long long)(uintptr_t)gsrc;
    u32x4 g0;
    g0[0] = 1u;                                        // count=1 (valid), user mode
    g0[1] = lds_off;                                   // lds_addr (bytes)
    g0[2] = (unsigned)(ga & 0xffffffffu);              // global_addr[31:0]
    g0[3] = (unsigned)((ga >> 32) & 0x01ffffffu)       // global_addr[56:32]
          | (2u << 30);                                // type = 2 ("image")
    i32x8 g1;
    g1[0] = (int)(0x00010000u                          // wg_mask=0, data_size=1 (2B)
          | (1u << 20)                                 // pad_enable
          | (5u << 22)                                 // pad_interval: 64 dwords (256B)
          | (3u << 25));                               // pad_amount: 4 dwords (16B)
    g1[1] = (int)(((unsigned)KTOT & 0xffffu) << 16);   // tensor_dim0[15:0] @ [63:48]
    g1[2] = (int)((((unsigned)KTOT >> 16) & 0xffffu)   // tensor_dim0[31:16]
          | (128u << 16));                             // tensor_dim1[15:0] = 128
    g1[3] = (int)(128u << 16);                         // tile_dim0 = 128 @ [127:112]
    g1[4] = 128;                                       // tile_dim1 = 128
    g1[5] = (int)KTOT;                                 // tensor_dim0_stride[31:0]
    g1[6] = 0;
    g1[7] = 0;
    i32x4 z4 = {0, 0, 0, 0};
#if defined(__clang_major__) && (__clang_major__ >= 23)
    i32x8 z8 = {0, 0, 0, 0, 0, 0, 0, 0};
    __builtin_amdgcn_tensor_load_to_lds(g0, g1, z4, z4, z8, 0);
#else
    __builtin_amdgcn_tensor_load_to_lds(g0, g1, z4, z4, 0);
#endif
}
#endif

// ---------------------------------------------------------------------------
// 4) fused masked-im2col GEMM: out[n][o][l] = Wp(128x3456) x B(3456x128-tile)
//    27 groups x 128-K steps; A tiles DMA'd by TDM (double-buffered, padded);
//    B staged as 8x b128 per thread from channel-last padded x; per group:
//    4 sub-chunks x 8 WMMAs per wave between one barrier pair.
// ---------------------------------------------------------------------------
__global__ __launch_bounds__(256)
void gemm_kernel(const __bf16* __restrict__ Wp, const __bf16* __restrict__ xpad,
                 const unsigned char* __restrict__ masks, float* __restrict__ out) {
    __shared__ __bf16 lds_a[2][128 * TSTR];  // A tiles (double buffer), row-major
    __shared__ __bf16 lds_b[128 * TSTR];     // B tile, column-major [col][k]

    const int tid   = threadIdx.x;
    const int n     = blockIdx.y;
    const int l0    = blockIdx.x * 128;
    const int lane  = tid & 31;
    const int wave  = tid >> 5;
    const int halfw = lane >> 4;             // 0: lanes 0-15, 1: lanes 16-31
    const int lrow  = lane & 15;
    const int mbase = (wave & 3) * 32;       // 4 wave-rows of 32 output channels
    const int nbase = (wave >> 2) * 64;      // 2 wave-cols of 64 pixels

    // B staging geometry: this thread owns column `col` and half the channels
    const int col   = tid & 127;             // pixel within tile
    const int kpart = tid >> 7;              // 0/1 -> channels 64*kpart..+63
    const int oh    = (l0 + col) >> 6;
    const int ow    = (l0 + col) & 63;

#if USE_TDM
    const unsigned lds_a_base = (unsigned)(uintptr_t)(&lds_a[0][0]);
#endif

    v8f acc[2][4];
#pragma unroll
    for (int mi = 0; mi < 2; ++mi)
#pragma unroll
        for (int ni = 0; ni < 4; ++ni)
#pragma unroll
            for (int r = 0; r < 8; ++r) acc[mi][ni][r] = 0.0f;

#if USE_TDM
    if (wave == 0) tdm_load_a_tile(Wp, lds_a_base);          // prefetch g=0 -> buf0
#endif

    for (int g = 0; g < NGRP; ++g) {         // 27 (kp,branch) groups
        const int buf = g & 1;
        const int kp  = g / 3;
        const int br  = g - kp * 3;
        const int dh  = kp / 3 - 1;
        const int dw  = kp - (kp / 3) * 3 - 1;

        // ---- stage B tile: one mask byte / pixel coord per group,
        //      8x b128 (64 contiguous channels) per thread ----
        {
            const unsigned char msk =
                masks[((br * NIMG + n) * KPP + kp) * LTOT + l0 + col];
            const int y  = oh + dh + 1;      // 0..65 in padded image
            const int x1 = ow + dw + 1;      // 0..65
            const v4u* src = (const v4u*)(xpad +
                (((size_t)n * H2 + y) * W2 + x1) * CIN + kpart * 64);
            v4u* dst = (v4u*)(lds_b + col * TSTR + kpart * 64);
            if (msk) {
#pragma unroll
                for (int i = 0; i < 8; ++i) dst[i] = src[i];
            } else {
                v4u z = {0u, 0u, 0u, 0u};
#pragma unroll
                for (int i = 0; i < 8; ++i) dst[i] = z;
            }
        }

#if USE_TDM
        if (wave == 0) {
            if (g + 1 < NGRP) {
                // prefetch next group's A tile into the other buffer
                tdm_load_a_tile(Wp + (size_t)(g + 1) * 128,
                                lds_a_base + (unsigned)((buf ^ 1) * ABUF_BYTES));
                __builtin_amdgcn_s_wait_tensorcnt(1);
            } else {
                __builtin_amdgcn_s_wait_tensorcnt(0);
            }
        }
#else
        // fallback: manual A staging (b128 loads) into the current buffer
#pragma unroll
        for (int i = 0; i < 8; ++i) {
            int chunk = tid + i * 256;       // 0..2047, 8 bf16 each
            int row   = chunk >> 4;
            int seg   = (chunk & 15) << 3;   // element offset in row
            *(v4u*)(&lds_a[buf][0] + row * TSTR + seg) =
                *(const v4u*)(Wp + (size_t)row * KTOT + g * 128 + seg);
        }
#endif
        __syncthreads();                     // B ready; A[buf] ready

        // ---- 4 sub-chunks of K=32: fragments + WMMA ----
#pragma unroll
        for (int ksub = 0; ksub < 4; ++ksub) {
            BFrag af[2];
#pragma unroll
            for (int mi = 0; mi < 2; ++mi) {
                // 16-bit A 16x32 layout: lanes 0-15 K=0..7,16..23;
                //                        lanes 16-31 K=8..15,24..31
                const __bf16* ap = &lds_a[buf][0] +
                    (mbase + mi * 16 + lrow) * TSTR + ksub * 32 + halfw * 8;
                af[mi].q[0] = *(const v4u*)(ap);
                af[mi].q[1] = *(const v4u*)(ap + 16);
            }
#pragma unroll
            for (int ni = 0; ni < 4; ++ni) {
                // 16-bit B 32x16 layout: lanes 0-15 K=0..7,16..23;
                //                        lanes 16-31 K=8..15,24..31
                BFrag bf;
                const __bf16* bp = lds_b +
                    (nbase + ni * 16 + lrow) * TSTR + ksub * 32 + halfw * 8;
                bf.q[0] = *(const v4u*)(bp);
                bf.q[1] = *(const v4u*)(bp + 16);
#pragma unroll
                for (int mi = 0; mi < 2; ++mi) {
                    acc[mi][ni] = __builtin_amdgcn_wmma_f32_16x16x32_bf16(
                        /*neg_a=*/false, af[mi].v, /*neg_b=*/false, bf.v,
                        /*c_mod=*/(short)0, acc[mi][ni],
                        /*reuse_a=*/false, /*reuse_b=*/false);
                }
            }
        }
        __syncthreads();                     // compute done before B is restaged
    }

    // ---- store: C/D layout VGPR r -> M = base + halfw*8 + r, N = lrow ----
#pragma unroll
    for (int mi = 0; mi < 2; ++mi) {
#pragma unroll
        for (int ni = 0; ni < 4; ++ni) {
            int rowb = mbase + mi * 16 + halfw * 8;
            int colw = l0 + nbase + ni * 16 + lrow;
            float* op = out + (n * OOUT + rowb) * LTOT + colw;
#pragma unroll
            for (int r = 0; r < 8; ++r)
                op[r * LTOT] = acc[mi][ni][r];
        }
    }
}

// ---------------------------------------------------------------------------
extern "C" void kernel_launch(void* const* d_in, const int* in_sizes, int n_in,
                              void* d_out, int out_size, void* d_ws, size_t ws_size,
                              hipStream_t stream) {
    (void)in_sizes; (void)n_in; (void)out_size; (void)ws_size;
    const float* x     = (const float*)d_in[0];
    const float* depth = (const float*)d_in[1];
    const float* fx    = (const float*)d_in[2];
    const float* w0    = (const float*)d_in[3];
    const float* w1    = (const float*)d_in[4];
    const float* w2    = (const float*)d_in[5];
    float* out = (float*)d_out;

    __bf16* Wp           = (__bf16*)((char*)d_ws + WP_OFF);
    __bf16* xpad         = (__bf16*)((char*)d_ws + XP_OFF);
    unsigned char* masks = (unsigned char*)d_ws + MK_OFF;

    const int npad = NIMG * CIN * H2 * W2;                     // 4,460,544
    pad_x_kernel<<<(npad + 255) / 256, 256, 0, stream>>>(x, xpad);
    pack_w_kernel<<<(OOUT * KTOT + 255) / 256, 256, 0, stream>>>(w0, w1, w2, Wp);
    mask_kernel<<<(NIMG * LTOT) / 256, 256, 0, stream>>>(depth, fx, masks);

    dim3 grid(LTOT / 128, NIMG);   // 32 x 8 = 256 workgroups
    gemm_kernel<<<grid, 256, 0, stream>>>(Wp, xpad, masks, out);
}
